// RegionProposalNetwork_1941325218125
// MI455X (gfx1250) — matrix operand order repaired and checked by reading, not compiled, not run on hardware
//
#include <hip/hip_runtime.h>
#include <hip/hip_bf16.h>
#include <math.h>

typedef __attribute__((ext_vector_type(16))) __bf16 v16bf;
typedef __attribute__((ext_vector_type(8)))  __bf16 v8bf;
typedef __attribute__((ext_vector_type(8)))  float  v8f;

#define H 50
#define W 50
#define NIMG 4
#define CIN 512
#define COUT 512
#define P (NIMG*H*W)          // 10000 spatial positions
#define KTOT (CIN*9)          // 4608
#define NANCH 22500           // 50*50*9
#define NPRE 3000
#define NPOST 300

// ---------------------------------------------------------------- pack x: NCHW f32 -> NHWC bf16
__global__ void pack_x_kernel(const float* __restrict__ x, __bf16* __restrict__ xbf) {
  int t = blockIdx.x * blockDim.x + threadIdx.x;
  if (t >= P * CIN) return;
  int c  = t & 511;
  int sp = t >> 9;            // n*2500 + y*50 + x
  int xx = sp % W;
  int yy = (sp / W) % H;
  int nn = sp / (H * W);
  xbf[t] = (__bf16)x[((nn * CIN + c) * H + yy) * W + xx];
}

// ---------------------------------------------------------------- pack w: OIHW f32 -> [k'/32][co][32] bf16, k' = (dy*3+dx)*512+ci
__global__ void pack_w_kernel(const float* __restrict__ w, __bf16* __restrict__ wpk) {
  int t = blockIdx.x * blockDim.x + threadIdx.x;
  if (t >= KTOT * COUT) return;
  int kk  = t & 31;
  int co  = (t >> 5) & 511;
  int blk = t >> 14;
  int kp  = blk * 32 + kk;
  int dd  = kp >> 9;          // dy*3+dx
  int ci  = kp & 511;
  int dy  = dd / 3, dx = dd % 3;
  wpk[t] = (__bf16)w[(co * CIN + ci) * 9 + dy * 3 + dx];
}

// ---------------------------------------------------------------- 3x3 conv + bias + relu
// Implicit GEMM, one wave computes a 16(spatial) x 64(cout) tile:
// A fragment (16x32 bf16) is reused by 4 WMMAs -> 4x less A traffic,
// filter-tap loop hoisted so bounds mask + addresses are K-loop invariant.
__global__ void __launch_bounds__(128)
conv3x3_wmma_kernel(const __bf16* __restrict__ xbf, const __bf16* __restrict__ wpk,
                    const float* __restrict__ bias, __bf16* __restrict__ featbf) {
  const int lane = threadIdx.x & 31;
  const int wv   = threadIdx.x >> 5;
  const int wave = blockIdx.x * 4 + wv;      // 0..4999
  const int mt = wave >> 3;                  // spatial tile 0..624
  const int ng = wave & 7;                   // cout group 0..7 (64 couts each)
  const int mbase  = mt * 16;
  const int cobase = ng * 64;
  const int hf = lane >> 4;                  // half-wave: selects K sub-range
  const int mr = lane & 15;                  // row within tile (A) / col (B,C,D)
  const int m  = mbase + mr;
  const int nn = m / (H * W);
  const int rem = m % (H * W);
  const int oy = rem / W;
  const int ox = rem % W;

  v8f acc[4];
  #pragma unroll
  for (int t = 0; t < 4; ++t) {
    const float b = bias[cobase + t * 16 + mr];
    #pragma unroll
    for (int r = 0; r < 8; ++r) acc[t][r] = b;
  }

  #pragma unroll
  for (int dd = 0; dd < 9; ++dd) {           // filter tap (dy,dx) -- constants after unroll
    const int dy = dd / 3, dx = dd % 3;
    const int iy = oy + dy - 1, ix = ox + dx - 1;
    const bool inb = ((unsigned)iy < (unsigned)H) && ((unsigned)ix < (unsigned)W);
    // A row base for this tap (per-lane); ci advances by 32 per step.
    const __bf16* abase = inb ? (xbf + (((size_t)(nn * H + iy) * W + ix) << 9) + hf * 16)
                              : xbf;
    // B base for this tap: blk = dd*16 + s ; elem ((blk*512)+co)*32 + kk
    const __bf16* wbase = wpk + ((size_t)(dd * 16) * COUT + (cobase + mr)) * 32 + hf * 16;

    #pragma unroll 4
    for (int s = 0; s < 16; ++s) {           // ci steps of 32
      v16bf a;
      #pragma unroll
      for (int e = 0; e < 16; ++e) a[e] = (__bf16)0.0f;
      if (inb) {
        const __bf16* src = abase + s * 32;
        v8bf lo = *(const v8bf*)(src);       // K = hf*16 + 0..7
        v8bf hi = *(const v8bf*)(src + 8);   // K = hf*16 + 8 + 0..7
        #pragma unroll
        for (int e = 0; e < 8; ++e) { a[e] = lo[e]; a[8 + e] = hi[e]; }
      }
      const __bf16* wp0 = wbase + (size_t)s * (COUT * 32);
      #pragma unroll
      for (int t = 0; t < 4; ++t) {
        v16bf bf = *(const v16bf*)(wp0 + t * (16 * 32));   // K = hf*16 + 0..15
        acc[t] = __builtin_amdgcn_wmma_f32_16x16x32_bf16(false, a, false, bf,
                                                         (short)0, acc[t], false, false);
      }
    }
  }

  #pragma unroll
  for (int t = 0; t < 4; ++t) {
    #pragma unroll
    for (int r = 0; r < 8; ++r) {
      float v = acc[t][r] > 0.f ? acc[t][r] : 0.f;   // ReLU
      int row = mbase + r + hf * 8;                  // D: lanes0-15 -> M=r, lanes16-31 -> M=r+8
      featbf[(size_t)row * COUT + cobase + t * 16 + mr] = (__bf16)v;
    }
  }
}

// ---------------------------------------------------------------- 1x1 heads: loc (36) + score (18)
__global__ void head_kernel(const __bf16* __restrict__ featbf,
                            const float* __restrict__ loc_w, const float* __restrict__ loc_b,
                            const float* __restrict__ score_w, const float* __restrict__ score_b,
                            float* __restrict__ out_loc, float* __restrict__ out_sco) {
  int t = blockIdx.x * blockDim.x + threadIdx.x;
  if (t >= P * 54) return;
  int p = t / 54, o = t % 54;
  const v8bf* fv = (const v8bf*)(featbf + (size_t)p * COUT);
  const float* wp; float bb;
  if (o < 36) { wp = loc_w + o * COUT; bb = loc_b[o]; }
  else        { wp = score_w + (o - 36) * COUT; bb = score_b[o - 36]; }
  float s = bb;
  for (int c8 = 0; c8 < COUT / 8; ++c8) {
    v8bf v = fv[c8];
    #pragma unroll
    for (int e = 0; e < 8; ++e) s += (float)v[e] * wp[c8 * 8 + e];
  }
  int n = p / (H * W), sp = p % (H * W);
  if (o < 36) out_loc[(size_t)n * (NANCH * 4) + sp * 36 + o] = s;
  else        out_sco[(size_t)n * (NANCH * 2) + sp * 18 + (o - 36)] = s;
}

// ---------------------------------------------------------------- anchors + decode + clip + min-size + softmax fg
__global__ void decode_kernel(const float* __restrict__ out_loc, const float* __restrict__ out_sco,
                              const int* __restrict__ img_h, const int* __restrict__ img_w,
                              float* __restrict__ boxes, float* __restrict__ score,
                              float* __restrict__ anchor_out) {
  int t = blockIdx.x * blockDim.x + threadIdx.x;
  if (t >= NIMG * NANCH) return;
  int n = t / NANCH, a = t % NANCH;
  int s = a / 9, ar = a % 9;
  int ay = s / W, ax = s % W;
  int ri = ar / 3, si = ar % 3;
  const float ratios[3] = {0.5f, 1.0f, 2.0f};
  const float scales[3] = {8.0f, 16.0f, 32.0f};
  float r  = ratios[ri], sc = scales[si];
  float hh = 16.0f * sc * sqrtf(r);
  float ww = 16.0f * sc * sqrtf(1.0f / r);
  float cx0 = ax * 16.0f + 8.0f, cy0 = ay * 16.0f + 8.0f;
  float x1a = cx0 - 0.5f * ww, y1a = cy0 - 0.5f * hh;
  float x2a = cx0 + 0.5f * ww, y2a = cy0 + 0.5f * hh;
  if (n == 0) {
    anchor_out[a * 4 + 0] = x1a; anchor_out[a * 4 + 1] = y1a;
    anchor_out[a * 4 + 2] = x2a; anchor_out[a * 4 + 3] = y2a;
  }
  const float* loc = out_loc + (size_t)n * (NANCH * 4) + a * 4;
  float aw = x2a - x1a, ah = y2a - y1a;
  float acx = x1a + 0.5f * aw, acy = y1a + 0.5f * ah;
  float cx = loc[0] * aw + acx, cy = loc[1] * ah + acy;
  float bw = expf(loc[2]) * aw, bh = expf(loc[3]) * ah;
  float imw = (float)img_w[0], imh = (float)img_h[0];
  float x1 = fminf(fmaxf(cx - 0.5f * bw, 0.f), imw);
  float y1 = fminf(fmaxf(cy - 0.5f * bh, 0.f), imh);
  float x2 = fminf(fmaxf(cx + 0.5f * bw, 0.f), imw);
  float y2 = fminf(fmaxf(cy + 0.5f * bh, 0.f), imh);
  const float* sco = out_sco + (size_t)n * (NANCH * 2) + a * 2;
  float mx = fmaxf(sco[0], sco[1]);
  float e0 = expf(sco[0] - mx), e1 = expf(sco[1] - mx);
  float fg = e1 / (e0 + e1);
  bool valid = (x2 - x1 + 1.0f >= 16.0f) && (y2 - y1 + 1.0f >= 16.0f);
  float sv = valid ? fg : -__builtin_inff();
  float* bb = boxes + ((size_t)n * NANCH + a) * 4;
  bb[0] = x1; bb[1] = y1; bb[2] = x2; bb[3] = y2;
  score[n * NANCH + a] = sv;
}

// ---------------------------------------------------------------- exact stable top-3000 via rank counting
__global__ void topk_kernel(const float* __restrict__ score, const float* __restrict__ boxes,
                            float* __restrict__ topbox, float* __restrict__ topsc) {
  int n = blockIdx.y;
  int i = blockIdx.x * 256 + threadIdx.x;
  const float* s = score + (size_t)n * NANCH;
  float si = (i < NANCH) ? s[i] : 0.f;
  int rank = 0;
  __shared__ float sh[256];
  for (int base = 0; base < NANCH; base += 256) {
    int j = base + threadIdx.x;
    sh[threadIdx.x] = (j < NANCH) ? s[j] : 0.f;
    __syncthreads();
    int lim = min(256, NANCH - base);
    if (i < NANCH) {
      for (int jj = 0; jj < lim; ++jj) {
        float sj = sh[jj];
        int j2 = base + jj;
        rank += ((sj > si) || (sj == si && j2 < i)) ? 1 : 0;
      }
    }
    __syncthreads();
  }
  if (i < NANCH && rank < NPRE) {
    topsc[n * NPRE + rank] = si;
    const float* bb = boxes + ((size_t)n * NANCH + i) * 4;
    float* tb = topbox + ((size_t)n * NPRE + rank) * 4;
    tb[0] = bb[0]; tb[1] = bb[1]; tb[2] = bb[2]; tb[3] = bb[3];
  }
}

// ---------------------------------------------------------------- greedy NMS + compact to 300 + roi_indices
__global__ void __launch_bounds__(512)
nms_kernel(const float* __restrict__ topbox, const float* __restrict__ topsc,
           float* __restrict__ rois, float* __restrict__ roi_idx) {
  int n = blockIdx.x;
  __shared__ float bx1[NPRE], by1[NPRE], bx2[NPRE], by2[NPRE];
  __shared__ int   keep[NPRE];
  for (int i = threadIdx.x; i < NPRE; i += blockDim.x) {
    const float* b = topbox + ((size_t)n * NPRE + i) * 4;
    bx1[i] = b[0]; by1[i] = b[1]; bx2[i] = b[2]; by2[i] = b[3];
    keep[i] = (topsc[n * NPRE + i] > -__builtin_inff()) ? 1 : 0;
  }
  __syncthreads();
  for (int i = 0; i < NPRE; ++i) {
    if (keep[i]) {
      float x1 = bx1[i], y1 = by1[i], x2 = bx2[i], y2 = by2[i];
      float ai = (x2 - x1) * (y2 - y1);
      for (int j = i + 1 + threadIdx.x; j < NPRE; j += blockDim.x) {
        float iw = fmaxf(fminf(x2, bx2[j]) - fmaxf(x1, bx1[j]), 0.f);
        float ih = fmaxf(fminf(y2, by2[j]) - fmaxf(y1, by1[j]), 0.f);
        float inter = iw * ih;
        float aj = (bx2[j] - bx1[j]) * (by2[j] - by1[j]);
        float iou = inter / (ai + aj - inter + 1e-9f);
        if (iou > 0.7f) keep[j] = 0;
      }
    }
    __syncthreads();
  }
  if (threadIdx.x == 0) {
    int cnt = 0;
    for (int i = 0; i < NPRE && cnt < NPOST; ++i) {
      if (keep[i]) {
        float* o = rois + ((size_t)n * NPOST + cnt) * 4;
        o[0] = bx1[i]; o[1] = by1[i]; o[2] = bx2[i]; o[3] = by2[i];
        cnt++;
      }
    }
    for (; cnt < NPOST; ++cnt) {
      float* o = rois + ((size_t)n * NPOST + cnt) * 4;
      o[0] = 0.f; o[1] = 0.f; o[2] = 0.f; o[3] = 0.f;
    }
  }
  for (int k = threadIdx.x; k < NPOST; k += blockDim.x)
    roi_idx[n * NPOST + k] = (float)n;
}

// ---------------------------------------------------------------- launch
extern "C" void kernel_launch(void* const* d_in, const int* in_sizes, int n_in,
                              void* d_out, int out_size, void* d_ws, size_t ws_size,
                              hipStream_t stream) {
  const float* x        = (const float*)d_in[0];
  const float* conv1_w  = (const float*)d_in[1];
  const float* conv1_b  = (const float*)d_in[2];
  const float* loc_w    = (const float*)d_in[3];
  const float* loc_b    = (const float*)d_in[4];
  const float* score_w  = (const float*)d_in[5];
  const float* score_b  = (const float*)d_in[6];
  const int*   img_h    = (const int*)d_in[7];
  const int*   img_w    = (const int*)d_in[8];

  float* out_loc   = (float*)d_out;                 // [4,22500,4]  360000
  float* out_sco   = out_loc + 360000;              // [4,22500,2]  180000
  float* out_rois  = out_sco + 180000;              // [1200,4]       4800
  float* out_ridx  = out_rois + 4800;               // [1200]         1200
  float* out_anch  = out_ridx + 1200;               // [22500,4]     90000

  char* ws = (char*)d_ws;
  size_t off = 0;
  __bf16* xbf    = (__bf16*)(ws + off); off += (size_t)P * CIN * 2;          // 10,240,000
  __bf16* wpk    = (__bf16*)(ws + off); off += (size_t)KTOT * COUT * 2;      //  4,718,592
  __bf16* featbf = (__bf16*)(ws + off); off += (size_t)P * COUT * 2;         // 10,240,000
  float*  fgbox  = (float*)(ws + off);  off += (size_t)NIMG * NANCH * 4 * 4; //  1,440,000
  float*  fgsco  = (float*)(ws + off);  off += (size_t)NIMG * NANCH * 4;     //    360,000
  float*  topbox = (float*)(ws + off);  off += (size_t)NIMG * NPRE * 4 * 4;  //    192,000
  float*  topsc  = (float*)(ws + off);  off += (size_t)NIMG * NPRE * 4;      //     48,000
  (void)ws_size; (void)in_sizes; (void)n_in; (void)out_size;

  pack_x_kernel<<<(P * CIN + 255) / 256, 256, 0, stream>>>(x, xbf);
  pack_w_kernel<<<(KTOT * COUT + 255) / 256, 256, 0, stream>>>(conv1_w, wpk);

  // 625 M-tiles * 8 N-groups = 5000 wave-tiles (16x64 each), 4 waves/block
  conv3x3_wmma_kernel<<<1250, 128, 0, stream>>>(xbf, wpk, conv1_b, featbf);

  head_kernel<<<(P * 54 + 255) / 256, 256, 0, stream>>>(featbf, loc_w, loc_b,
                                                        score_w, score_b, out_loc, out_sco);

  decode_kernel<<<(NIMG * NANCH + 255) / 256, 256, 0, stream>>>(out_loc, out_sco, img_h, img_w,
                                                                fgbox, fgsco, out_anch);

  dim3 tg((NANCH + 255) / 256, NIMG);
  topk_kernel<<<tg, 256, 0, stream>>>(fgsco, fgbox, topbox, topsc);

  nms_kernel<<<NIMG, 512, 0, stream>>>(topbox, topsc, out_rois, out_ridx);
}
